// GraphEncoder_18992345383330
// MI455X (gfx1250) — compile-verified, matrix operand order checked by reading, not run on hardware
//
#include <hip/hip_runtime.h>
#include <hip/hip_bf16.h>

typedef __attribute__((ext_vector_type(16))) _Float16 v16h;
typedef __attribute__((ext_vector_type(8)))  _Float16 v8h;
typedef __attribute__((ext_vector_type(8)))  float    v8f;

#define NN 4096
#define EE 131072
#define HIDD 256
#define OUTD 512
#define SCALE_QK 0.125f
#define MASKF (-10000.0f)

// ---------------------------------------------------------------------------
// WMMA helpers (CDNA5 16x16x32 f16 -> f32)
// Fragment packing (16-bit A / B, wave32):
//   lane L: row = L&15 ; klo = (L&16)?8:0 ; halves = T[row][klo+0..7] , T[row][klo+16..23]
// Both A (row-major [16,K]) and B (row-major [N=16,K], i.e. C = A * B^T) use
// the identical load pattern.
// ---------------------------------------------------------------------------
__device__ __forceinline__ v16h load_frag(const _Float16* __restrict__ p, int ld) {
  int lane = threadIdx.x & 31;
  int r    = lane & 15;
  int klo  = (lane & 16) ? 8 : 0;
  const _Float16* q = p + (size_t)r * ld + klo;
  v8h lo = *(const v8h*)(q);
  v8h hi = *(const v8h*)(q + 16);
  v16h out;
#pragma unroll
  for (int i = 0; i < 8; ++i) { out[i] = lo[i]; out[i + 8] = hi[i]; }
  return out;
}

__device__ __forceinline__ v8f wmma_f16f32(v16h a, v16h b, v8f c) {
  return __builtin_amdgcn_wmma_f32_16x16x32_f16(false, a, false, b, (short)0, c,
                                                false, false);
}

// ---------------------------------------------------------------------------
// Generic GEMM: C[M,N] = A[M,K] * B[N,K]^T + bias[N]
// A,B f16 row-major. Outputs: optional f32 C, optional f16 C, optional f16 C^T.
// Block = 128 threads (4 waves); block tile 64x64; wave tile 16x64.
// M,N multiples of 64; K multiple of 32.
// ---------------------------------------------------------------------------
__global__ __launch_bounds__(128) void gemm_f16(
    const _Float16* __restrict__ A, const _Float16* __restrict__ B,
    const float* __restrict__ bias, int M, int N, int K,
    float* __restrict__ Cf, _Float16* __restrict__ Ch, _Float16* __restrict__ Ct) {
  int w    = threadIdx.x >> 5;
  int lane = threadIdx.x & 31;
  int n    = lane & 15;
  int mb   = (lane >> 4) * 8;
  int row0 = blockIdx.x * 64 + w * 16;
  int col0 = blockIdx.y * 64;

  v8f acc[4] = {};
  for (int kb = 0; kb < K; kb += 32) {
    v16h a = load_frag(A + (size_t)row0 * K + kb, K);
#pragma unroll
    for (int j = 0; j < 4; ++j) {
      v16h b = load_frag(B + (size_t)(col0 + 16 * j) * K + kb, K);
      acc[j] = wmma_f16f32(a, b, acc[j]);
    }
  }
#pragma unroll
  for (int j = 0; j < 4; ++j) {
    int col  = col0 + 16 * j + n;
    float bv = bias ? bias[col] : 0.0f;
#pragma unroll
    for (int r = 0; r < 8; ++r) {
      float v = acc[j][r] + bv;
      int row = row0 + mb + r;
      if (Cf) Cf[(size_t)row * N + col] = v;
      if (Ch) Ch[(size_t)row * N + col] = (_Float16)v;
      if (Ct) Ct[(size_t)col * M + row] = (_Float16)v;
    }
  }
}

// ---------------------------------------------------------------------------
// GAT attention, streaming online-softmax.
// Block = 128 threads = 4 waves; wave w = head w; 16 query rows per block.
// Keys streamed in 32-wide tiles. S tile staged in LDS for adj/mask/edge fix-up.
// ---------------------------------------------------------------------------
__global__ __launch_bounds__(128) void gat_attention(
    const _Float16* __restrict__ Qh, const _Float16* __restrict__ Kh,
    const _Float16* __restrict__ Vt, const float* __restrict__ adj,
    const int* __restrict__ row_start, const int* __restrict__ csr_eid,
    const int* __restrict__ etgt, const float* __restrict__ eattn,
    _Float16* __restrict__ outh) {
  __shared__ alignas(16) float    sS[4][16][32];
  __shared__ alignas(16) _Float16 pS[4][16][32];
  __shared__ alignas(16) float    sAdj[16][32];
  __shared__ float sFac[4][16];
  __shared__ float sSum[4][16];

  int tid   = threadIdx.x;
  int w     = tid >> 5;       // head
  int lane  = tid & 31;
  int n     = lane & 15;
  int mb    = (lane >> 4) * 8;
  int hd0   = w * 64;
  int qbase = blockIdx.x * 16;

  v16h aQ0 = load_frag(Qh + (size_t)qbase * HIDD + hd0, HIDD);
  v16h aQ1 = load_frag(Qh + (size_t)qbase * HIDD + hd0 + 32, HIDD);

  v8f accO[4] = {};
  float rmax = -3.0e38f, rsum = 0.0f;
  int r = lane;  // lanes 0..15 own one query row each for scalar phase
  int estart = 0, eend = 0;
  if (lane < 16) { estart = row_start[qbase + r]; eend = row_start[qbase + r + 1]; }

  for (int kb = 0; kb < NN; kb += 32) {
    // S = Q(16x64) * K_tile^T(64x32)
#pragma unroll
    for (int j = 0; j < 2; ++j) {
      v8f acc = {};
      v16h b0 = load_frag(Kh + (size_t)(kb + 16 * j) * HIDD + hd0, HIDD);
      acc = wmma_f16f32(aQ0, b0, acc);
      v16h b1 = load_frag(Kh + (size_t)(kb + 16 * j) * HIDD + hd0 + 32, HIDD);
      acc = wmma_f16f32(aQ1, b1, acc);
#pragma unroll
      for (int rr = 0; rr < 8; ++rr) sS[w][mb + rr][16 * j + n] = acc[rr];
    }
    // stage adj tile once for all 4 heads (128 threads x 4 values)
#pragma unroll
    for (int q4 = 0; q4 < 4; ++q4) {
      int idx = q4 * 128 + tid;            // 0..511
      int rr = idx >> 5, cc = idx & 31;
      sAdj[rr][cc] = adj[(size_t)(qbase + rr) * NN + kb + cc];
    }
    __syncthreads();

    if (lane < 16) {
      int qr = qbase + r;
      // scale + adjacency bonus + mask
      for (int c = 0; c < 32; ++c) {
        float a = sAdj[r][c];
        float s = sS[w][r][c] * SCALE_QK + 10.0f * a;
        bool allowed = (qr == kb + c) || (a > 1e-8f);
        sS[w][r][c] = allowed ? s : MASKF;
      }
      // sparse per-edge attention bias (CSR by src)
      for (int jj = estart; jj < eend; ++jj) {
        int e = csr_eid[jj];
        int t = etgt[e] - kb;
        if ((unsigned)t < 32u) sS[w][r][t] += eattn[e * 4 + w];
      }
      // online softmax bookkeeping
      float tmax = -3.0e38f;
      for (int c = 0; c < 32; ++c) tmax = fmaxf(tmax, sS[w][r][c]);
      float nmax = fmaxf(rmax, tmax);
      float fac  = __expf(rmax - nmax);
      float psum = 0.0f;
      for (int c = 0; c < 32; ++c) {
        float p = __expf(sS[w][r][c] - nmax);
        pS[w][r][c] = (_Float16)p;
        psum += p;
      }
      rsum = rsum * fac + psum;
      rmax = nmax;
      sFac[w][r] = fac;
    }
    __syncthreads();

    float fr[8];
#pragma unroll
    for (int rr = 0; rr < 8; ++rr) fr[rr] = sFac[w][mb + rr];
    v16h aP = load_frag(&pS[w][0][0], 32);
#pragma unroll
    for (int d = 0; d < 4; ++d) {
#pragma unroll
      for (int rr = 0; rr < 8; ++rr) accO[d][rr] *= fr[rr];
      v16h bV = load_frag(Vt + (size_t)(hd0 + 16 * d) * NN + kb, NN);
      accO[d] = wmma_f16f32(aP, bV, accO[d]);
    }
    __syncthreads();
  }

  if (lane < 16) sSum[w][r] = rsum;
  __syncthreads();
  float inv[8];
#pragma unroll
  for (int rr = 0; rr < 8; ++rr) inv[rr] = 1.0f / sSum[w][mb + rr];
#pragma unroll
  for (int d = 0; d < 4; ++d)
#pragma unroll
    for (int rr = 0; rr < 8; ++rr) {
      int row = qbase + mb + rr;
      outh[(size_t)row * HIDD + hd0 + 16 * d + n] = (_Float16)(accO[d][rr] * inv[rr]);
    }
}

// ---------------------------------------------------------------------------
// Elementwise / small kernels
// ---------------------------------------------------------------------------
__global__ void pad_cvt_f16(const float* __restrict__ src, _Float16* __restrict__ dst,
                            int rows, int scols, int dcols) {
  int idx = blockIdx.x * blockDim.x + threadIdx.x;
  if (idx >= rows * dcols) return;
  int rr = idx / dcols, cc = idx - rr * dcols;
  dst[idx] = (cc < scols) ? (_Float16)src[(size_t)rr * scols + cc] : (_Float16)0.0f;
}

__global__ void cvt_f16(const float* __restrict__ src, _Float16* __restrict__ dst, int n) {
  int i = blockIdx.x * blockDim.x + threadIdx.x;
  if (i < n) dst[i] = (_Float16)src[i];
}

// row-wise LayerNorm (+GELU) over HIDD=256, one row per 256-thread block
__global__ __launch_bounds__(256) void ln_gelu_f16(
    const float* __restrict__ X, const float* __restrict__ g,
    const float* __restrict__ b, _Float16* __restrict__ Yh) {
  __shared__ float red[256];
  int row = blockIdx.x, t = threadIdx.x;
  float v = X[(size_t)row * HIDD + t];
  red[t] = v; __syncthreads();
  for (int off = 128; off > 0; off >>= 1) { if (t < off) red[t] += red[t + off]; __syncthreads(); }
  float m = red[0] / 256.0f; __syncthreads();
  float d = v - m;
  red[t] = d * d; __syncthreads();
  for (int off = 128; off > 0; off >>= 1) { if (t < off) red[t] += red[t + off]; __syncthreads(); }
  float var = red[0] / 256.0f;
  float y = d * rsqrtf(var + 1e-5f) * g[t] + b[t];
  float ge = 0.5f * y * (1.0f + erff(y * 0.70710678118f));
  Yh[(size_t)row * HIDD + t] = (_Float16)ge;
}

// x_new = LN(o + x) ; writes f32 (in-place ok) and f16 copy
__global__ __launch_bounds__(256) void residual_ln(
    const float* __restrict__ O, const float* __restrict__ Xin,
    const float* __restrict__ g, const float* __restrict__ b,
    float* __restrict__ Xout, _Float16* __restrict__ Xh) {
  __shared__ float red[256];
  int row = blockIdx.x, t = threadIdx.x;
  float v = O[(size_t)row * HIDD + t] + Xin[(size_t)row * HIDD + t];
  red[t] = v; __syncthreads();
  for (int off = 128; off > 0; off >>= 1) { if (t < off) red[t] += red[t + off]; __syncthreads(); }
  float m = red[0] / 256.0f; __syncthreads();
  float d = v - m;
  red[t] = d * d; __syncthreads();
  for (int off = 128; off > 0; off >>= 1) { if (t < off) red[t] += red[t + off]; __syncthreads(); }
  float var = red[0] / 256.0f;
  float y = d * rsqrtf(var + 1e-5f) * g[t] + b[t];
  Xout[(size_t)row * HIDD + t] = y;
  Xh[(size_t)row * HIDD + t] = (_Float16)y;
}

__global__ void zero_ints(int* __restrict__ p, int n) {
  int i = blockIdx.x * blockDim.x + threadIdx.x;
  if (i < n) p[i] = 0;
}
__global__ void hist_src(const int* __restrict__ src, int* __restrict__ counts, int E) {
  int e = blockIdx.x * blockDim.x + threadIdx.x;
  if (e < E) atomicAdd(&counts[src[e]], 1);
}
__global__ __launch_bounds__(1024) void scan4096(
    const int* __restrict__ counts, int* __restrict__ row_start, int* __restrict__ cursor) {
  __shared__ int tsum[1024];
  int t = threadIdx.x;
  int base = t * 4;
  int c0 = counts[base], c1 = counts[base + 1], c2 = counts[base + 2], c3 = counts[base + 3];
  int s = c0 + c1 + c2 + c3;
  tsum[t] = s; __syncthreads();
  for (int off = 1; off < 1024; off <<= 1) {
    int v = (t >= off) ? tsum[t - off] : 0;
    __syncthreads();
    tsum[t] += v;
    __syncthreads();
  }
  int excl = tsum[t] - s;
  row_start[base] = excl;     cursor[base] = excl;     excl += c0;
  row_start[base + 1] = excl; cursor[base + 1] = excl; excl += c1;
  row_start[base + 2] = excl; cursor[base + 2] = excl; excl += c2;
  row_start[base + 3] = excl; cursor[base + 3] = excl;
  if (t == 1023) row_start[4096] = tsum[t];
}
__global__ void scatter_csr(const int* __restrict__ src, int* __restrict__ cursor,
                            int* __restrict__ eid, int E) {
  int e = blockIdx.x * blockDim.x + threadIdx.x;
  if (e < E) { int p = atomicAdd(&cursor[src[e]], 1); eid[p] = e; }
}

// eattn[e][h] = ef[e,:] . We[h,:] + be[h]
__global__ __launch_bounds__(256) void edge_attn_k(
    const float* __restrict__ ef, const float* __restrict__ We,
    const float* __restrict__ be, float* __restrict__ eattn, int E) {
  __shared__ float sW[4][64];
  __shared__ float sb[4];
  int t = threadIdx.x;
  if (t < 256) sW[t >> 6][t & 63] = We[t];
  if (t < 4) sb[t] = be[t];
  __syncthreads();
  int e = blockIdx.x * 256 + t;
  if (e >= E) return;
  float a0 = sb[0], a1 = sb[1], a2 = sb[2], a3 = sb[3];
  const float* f = ef + (size_t)e * 64;
  for (int k = 0; k < 64; ++k) {
    float x = f[k];
    a0 += x * sW[0][k]; a1 += x * sW[1][k]; a2 += x * sW[2][k]; a3 += x * sW[3][k];
  }
  float* o = eattn + (size_t)e * 4;
  o[0] = a0; o[1] = a1; o[2] = a2; o[3] = a3;
}

__global__ __launch_bounds__(256) void colmean(const float* __restrict__ X,
                                               float* __restrict__ gv) {
  int t = threadIdx.x;
  float s = 0.0f;
  for (int i = 0; i < NN; ++i) s += X[(size_t)i * HIDD + t];
  gv[t] = s * (1.0f / (float)NN);
}

__global__ __launch_bounds__(512) void pool_head(
    const float* __restrict__ gvec, const float* __restrict__ W1,
    const float* __restrict__ b1, const float* __restrict__ g1,
    const float* __restrict__ bb1, const float* __restrict__ W2,
    const float* __restrict__ b2, float* __restrict__ out) {
  __shared__ float sg[256], t1[256], red[256], sh2[256];
  int t = threadIdx.x;
  if (t < 256) sg[t] = gvec[t];
  __syncthreads();
  if (t < 256) {
    float a = b1[t];
    for (int k = 0; k < 256; ++k) a += W1[(size_t)t * 256 + k] * sg[k];
    t1[t] = a;
  }
  __syncthreads();
  if (t < 256) red[t] = t1[t];
  __syncthreads();
  for (int off = 128; off > 0; off >>= 1) { if (t < off) red[t] += red[t + off]; __syncthreads(); }
  float m = red[0] / 256.0f; __syncthreads();
  if (t < 256) { float d = t1[t] - m; red[t] = d * d; }
  __syncthreads();
  for (int off = 128; off > 0; off >>= 1) { if (t < off) red[t] += red[t + off]; __syncthreads(); }
  float var = red[0] / 256.0f; __syncthreads();
  if (t < 256) {
    float y = (t1[t] - m) * rsqrtf(var + 1e-5f) * g1[t] + bb1[t];
    sh2[t] = 0.5f * y * (1.0f + erff(y * 0.70710678118f));
  }
  __syncthreads();
  float a = b2[t];
  for (int k = 0; k < 256; ++k) a += W2[(size_t)t * 256 + k] * sh2[k];
  out[t] = a;
}

// ---------------------------------------------------------------------------
// Host launch
// ---------------------------------------------------------------------------
extern "C" void kernel_launch(void* const* d_in, const int* in_sizes, int n_in,
                              void* d_out, int out_size, void* d_ws, size_t ws_size,
                              hipStream_t stream) {
  (void)in_sizes; (void)n_in; (void)out_size; (void)ws_size;
  // Top-level in setup_inputs() insertion order; `params` leaves in JAX pytree
  // (sorted-key) order.
  const float* nf   = (const float*)d_in[0];
  const float* adj  = (const float*)d_in[1];
  const float* ef   = (const float*)d_in[2];
  const int*   esrc = (const int*)d_in[3];
  const int*   etgt = (const int*)d_in[4];
  // gat layer leaves (sorted): We, Wk, Wo, Wq, Wv, b, be, bk, bo, bq, bv, g
  const int GAT0 = 5;
  enum { iWe = 0, iWk, iWo, iWq, iWv, ib, ibe, ibk, ibo, ibq, ibv, ig };
  const int NE0 = GAT0 + 36;  // node_enc (sorted): W1, W2, b1, b2, bb1, g1
  const int PL0 = NE0 + 6;    // pool     (sorted): W1, W2, b1, b2, bb1, g1
  const float* neW1 = (const float*)d_in[NE0 + 0];
  const float* neW2 = (const float*)d_in[NE0 + 1];
  const float* neb1 = (const float*)d_in[NE0 + 2];
  const float* neb2 = (const float*)d_in[NE0 + 3];
  const float* nebb1= (const float*)d_in[NE0 + 4];
  const float* neg1 = (const float*)d_in[NE0 + 5];
  const float* pW1  = (const float*)d_in[PL0 + 0];
  const float* pW2  = (const float*)d_in[PL0 + 1];
  const float* pb1  = (const float*)d_in[PL0 + 2];
  const float* pb2  = (const float*)d_in[PL0 + 3];
  const float* pbb1 = (const float*)d_in[PL0 + 4];
  const float* pg1  = (const float*)d_in[PL0 + 5];

  // bump allocator over workspace
  char* wsb = (char*)d_ws;
  size_t off = 0;
  auto alloc = [&](size_t bytes) -> void* {
    void* p = wsb + off;
    off += (bytes + 255) & ~(size_t)255;
    return p;
  };
  _Float16* nfP   = (_Float16*)alloc((size_t)NN * 160 * 2);
  _Float16* W1P   = (_Float16*)alloc((size_t)256 * 160 * 2);
  _Float16* W2e16 = (_Float16*)alloc((size_t)65536 * 2);
  _Float16* Wq16  = (_Float16*)alloc((size_t)65536 * 2);
  _Float16* Wk16  = (_Float16*)alloc((size_t)65536 * 2);
  _Float16* Wv16  = (_Float16*)alloc((size_t)65536 * 2);
  _Float16* Wo16  = (_Float16*)alloc((size_t)65536 * 2);
  float*    hbuf  = (float*)alloc((size_t)NN * HIDD * 4);   // also reused as `o`
  _Float16* he16  = (_Float16*)alloc((size_t)NN * HIDD * 2);
  float*    x     = (float*)alloc((size_t)NN * HIDD * 4);
  _Float16* xh    = (_Float16*)alloc((size_t)NN * HIDD * 2);
  _Float16* Qh    = (_Float16*)alloc((size_t)NN * HIDD * 2);
  _Float16* Kh    = (_Float16*)alloc((size_t)NN * HIDD * 2);
  _Float16* Vt    = (_Float16*)alloc((size_t)HIDD * NN * 2);
  _Float16* aOut  = (_Float16*)alloc((size_t)NN * HIDD * 2);
  float*    eattn = (float*)alloc((size_t)EE * 4 * 4);
  int*      counts= (int*)alloc((size_t)NN * 4);
  int*      cursor= (int*)alloc((size_t)NN * 4);
  int*      rowst = (int*)alloc((size_t)(NN + 1) * 4);
  int*      eid   = (int*)alloc((size_t)EE * 4);
  float*    gvec  = (float*)alloc(256 * 4);

  auto gemm = [&](const _Float16* A, const _Float16* B, const float* bias,
                  int M, int N, int K, float* Cf, _Float16* Ch, _Float16* Ct) {
    dim3 grid(M / 64, N / 64);
    gemm_f16<<<grid, 128, 0, stream>>>(A, B, bias, M, N, K, Cf, Ch, Ct);
  };

  // ---- node encoder ----
  pad_cvt_f16<<<(NN * 160 + 255) / 256, 256, 0, stream>>>(nf, nfP, NN, 131, 160);
  pad_cvt_f16<<<(256 * 160 + 255) / 256, 256, 0, stream>>>(neW1, W1P, 256, 131, 160);
  cvt_f16<<<(65536 + 255) / 256, 256, 0, stream>>>(neW2, W2e16, 65536);
  gemm(nfP, W1P, neb1, NN, HIDD, 160, hbuf, nullptr, nullptr);
  ln_gelu_f16<<<NN, 256, 0, stream>>>(hbuf, neg1, nebb1, he16);
  gemm(he16, W2e16, neb2, NN, HIDD, 256, x, xh, nullptr);

  // ---- CSR by src for edge bias scatter ----
  zero_ints<<<(NN + 255) / 256, 256, 0, stream>>>(counts, NN);
  hist_src<<<(EE + 255) / 256, 256, 0, stream>>>(esrc, counts, EE);
  scan4096<<<1, 1024, 0, stream>>>(counts, rowst, cursor);
  scatter_csr<<<(EE + 255) / 256, 256, 0, stream>>>(esrc, cursor, eid, EE);

  // ---- GAT layers ----
  for (int l = 0; l < 3; ++l) {
    const float* We = (const float*)d_in[GAT0 + 12 * l + iWe];
    const float* Wk = (const float*)d_in[GAT0 + 12 * l + iWk];
    const float* Wo = (const float*)d_in[GAT0 + 12 * l + iWo];
    const float* Wq = (const float*)d_in[GAT0 + 12 * l + iWq];
    const float* Wv = (const float*)d_in[GAT0 + 12 * l + iWv];
    const float* lb = (const float*)d_in[GAT0 + 12 * l + ib];
    const float* be = (const float*)d_in[GAT0 + 12 * l + ibe];
    const float* bk = (const float*)d_in[GAT0 + 12 * l + ibk];
    const float* bo = (const float*)d_in[GAT0 + 12 * l + ibo];
    const float* bq = (const float*)d_in[GAT0 + 12 * l + ibq];
    const float* bv = (const float*)d_in[GAT0 + 12 * l + ibv];
    const float* lg = (const float*)d_in[GAT0 + 12 * l + ig];

    cvt_f16<<<(65536 + 255) / 256, 256, 0, stream>>>(Wq, Wq16, 65536);
    cvt_f16<<<(65536 + 255) / 256, 256, 0, stream>>>(Wk, Wk16, 65536);
    cvt_f16<<<(65536 + 255) / 256, 256, 0, stream>>>(Wv, Wv16, 65536);
    cvt_f16<<<(65536 + 255) / 256, 256, 0, stream>>>(Wo, Wo16, 65536);

    gemm(xh, Wq16, bq, NN, HIDD, 256, nullptr, Qh, nullptr);
    gemm(xh, Wk16, bk, NN, HIDD, 256, nullptr, Kh, nullptr);
    gemm(xh, Wv16, bv, NN, HIDD, 256, nullptr, nullptr, Vt);  // transposed V

    edge_attn_k<<<(EE + 255) / 256, 256, 0, stream>>>(ef, We, be, eattn, EE);

    gat_attention<<<NN / 16, 128, 0, stream>>>(Qh, Kh, Vt, adj, rowst, eid, etgt,
                                               eattn, aOut);

    gemm(aOut, Wo16, bo, NN, HIDD, 256, hbuf, nullptr, nullptr);
    residual_ln<<<NN, 256, 0, stream>>>(hbuf, x, lg, lb, x, xh);
  }

  // ---- pooling head ----
  colmean<<<1, 256, 0, stream>>>(x, gvec);
  pool_head<<<1, 512, 0, stream>>>(gvec, pW1, pb1, pg1, pbb1, pW2, pb2,
                                   (float*)d_out);
}
